// GATv2_47150150975825
// MI455X (gfx1250) — matrix operand order repaired and checked by reading, not compiled, not run on hardware
//
#include <hip/hip_runtime.h>
#include <hip/hip_bf16.h>

#define Bb 4
#define Nn 512
#define Hh 8

typedef float v2f  __attribute__((ext_vector_type(2)));
typedef float v8f  __attribute__((ext_vector_type(8)));
typedef __bf16 v8bf  __attribute__((ext_vector_type(8)));
typedef __bf16 v16bf __attribute__((ext_vector_type(16)));

__device__ __forceinline__ v8f wmma_f32(v2f a, v2f b, v8f c) {
  // V_WMMA_F32_16X16X4_F32 : exact fp32 path for the small GEMMs
  return __builtin_amdgcn_wmma_f32_16x16x4_f32(false, a, false, b, (short)0, c,
                                               false, false);
}
__device__ __forceinline__ v8f wmma_bf16(v16bf a, v16bf b, v8f c) {
  // V_WMMA_F32_16X16X32_BF16 : fast 16-bit XDL path, f32 accumulate
  return __builtin_amdgcn_wmma_f32_16x16x32_bf16(false, a, false, b, (short)0, c,
                                                 false, false);
}

// ---------------------------------------------------------------------------
// Kernel 0: graph projection  G[b,c] = bg[c] + sum_k gfts[b,k] * Wg[k,c]
// ---------------------------------------------------------------------------
__global__ void graph_proj_kernel(const float* __restrict__ gfts,
                                  const float* __restrict__ Wg,
                                  const float* __restrict__ bg,
                                  float* __restrict__ G) {
  int idx = blockIdx.x * 256 + threadIdx.x;
  if (idx >= Bb * 128) return;
  int b = idx >> 7, c = idx & 127;
  float acc = bg[c];
  for (int k = 0; k < 128; ++k) acc += gfts[b * 128 + k] * Wg[k * 128 + c];
  G[idx] = acc;
}

// ---------------------------------------------------------------------------
// Kernel 0b: split We (128x128 f32) into hi/lo bf16 planes, pre-packed in the
// WMMA B-fragment layout:  idx(kblk, col, half, e)  ->  We[kblk*32+8*half+e+(e>=8?8:0)][col]
// Each lane of the hot loop then reads 16 contiguous bf16 (32 B).
// ---------------------------------------------------------------------------
__global__ void we_pack_kernel(const float* __restrict__ We,
                               __bf16* __restrict__ WeHi,
                               __bf16* __restrict__ WeLo) {
  int idx = blockIdx.x * 256 + threadIdx.x;   // 4*128*2*16 = 16384
  if (idx >= 16384) return;
  int e    = idx & 15;
  int half = (idx >> 4) & 1;
  int c    = (idx >> 5) & 127;
  int kblk = idx >> 12;
  int k = kblk * 32 + 8 * half + e + (e >= 8 ? 8 : 0);
  float w = We[k * 128 + c];
  __bf16 hi = (__bf16)w;
  __bf16 lo = (__bf16)(w - (float)hi);
  WeHi[idx] = hi;
  WeLo[idx] = lo;
}

// ---------------------------------------------------------------------------
// Kernel 1: z-projections (fp32 WMMA, tiny fraction of total FLOPs).
// ---------------------------------------------------------------------------
__global__ void proj_z_kernel(const float* __restrict__ node,
                              const float* __restrict__ hidden,
                              const float* __restrict__ Wm, const float* __restrict__ bm,
                              const float* __restrict__ Ws, const float* __restrict__ bs,
                              const float* __restrict__ W1, const float* __restrict__ b1,
                              const float* __restrict__ W2, const float* __restrict__ b2,
                              float* __restrict__ V,  float* __restrict__ S,
                              float* __restrict__ P1, float* __restrict__ P2) {
  __shared__ float zt[16][257];
  const int tid = threadIdx.x;
  const int rb  = blockIdx.x * 16;

  for (int idx = tid; idx < 16 * 256; idx += 256) {
    int r = idx >> 8, c = idx & 255;
    int grow = rb + r;
    int b = grow >> 9, n = grow & 511;
    float v = (c < 128) ? node[(b * Nn + n) * 128 + c]
                        : hidden[(b * Nn + n) * 128 + (c - 128)];
    zt[r][c] = v;
  }
  __syncthreads();

  const int wave = tid >> 5, lane = tid & 31;
  const int nloc = lane & 15, half = lane >> 4;
  const int colb = wave * 16;

  const float* Wmat[4] = {Wm, Ws, W1, W2};
  const float* Bias[4] = {bm, bs, b1, b2};
  float*       Outp[4] = {V,  S,  P1, P2};

  for (int m = 0; m < 4; ++m) {
    const float* W = Wmat[m];
    v8f c = {};
    for (int k = 0; k < 256; k += 4) {
      int ka = k + half * 2;
      v2f a, bf;
      a.x  = zt[nloc][ka];
      a.y  = zt[nloc][ka + 1];
      bf.x = W[ka * 128 + colb + nloc];
      bf.y = W[(ka + 1) * 128 + colb + nloc];
      c = wmma_f32(a, bf, c);
    }
    float bv = Bias[m][colb + nloc];
    float* O = Outp[m];
#pragma unroll
    for (int r = 0; r < 8; ++r) {
      int grow = rb + r + 8 * half;
      O[grow * 128 + colb + nloc] = c[r] + bv;
    }
  }
}

// ---------------------------------------------------------------------------
// Kernel 2 (hot): fused edge GEMM (3-product bf16 split, f32 accumulate)
// + broadcast adds + leaky-relu + per-head Aw reduction -> logits.
// Block = (b, i, 16-wide j tile); wave h owns head h.
// ---------------------------------------------------------------------------
__global__ void edge_logits_kernel(const float* __restrict__ edge,
                                   const __bf16* __restrict__ WeHi,
                                   const __bf16* __restrict__ WeLo,
                                   const float* __restrict__ be,
                                   const float* __restrict__ P1,
                                   const float* __restrict__ P2,
                                   const float* __restrict__ G,
                                   const float* __restrict__ Aw,
                                   float* __restrict__ logits) {
  __shared__ __align__(16) __bf16 ehi[16][136];  // 68 dwords/row: conflict-free
  __shared__ __align__(16) __bf16 elo[16][136];
  __shared__ float p1t[16][129];
  __shared__ float radd[128];

  const int tid = threadIdx.x;
  const int jt = blockIdx.x, i = blockIdx.y, b = blockIdx.z;
  const int jb = jt * 16;

  const float* ebase = edge + ((size_t)(b * Nn + i) * Nn + jb) * 128;
  for (int idx = tid; idx < 16 * 128; idx += 256) {
    int r = idx >> 7, c = idx & 127;
    float v = ebase[(size_t)r * 128 + c];
    __bf16 hi = (__bf16)v;
    ehi[r][c] = hi;
    elo[r][c] = (__bf16)(v - (float)hi);
    p1t[r][c] = P1[(b * Nn + jb + r) * 128 + c];
  }
  if (tid < 128)
    radd[tid] = P2[(b * Nn + i) * 128 + tid] + G[b * 128 + tid] + be[tid];
  __syncthreads();

  const int h = tid >> 5, lane = tid & 31;
  const int nloc = lane & 15, half = lane >> 4;
  const int colb = h * 16;

  v8f c = {};
#pragma unroll
  for (int kblk = 0; kblk < 4; ++kblk) {
    const int kb = kblk * 32;
    // A fragments: lane (M=nloc, half) needs K = kb + 16*g + 8*half + r
    v8bf a0h = *(const v8bf*)&ehi[nloc][kb + 8 * half];
    v8bf a1h = *(const v8bf*)&ehi[nloc][kb + 16 + 8 * half];
    v8bf a0l = *(const v8bf*)&elo[nloc][kb + 8 * half];
    v8bf a1l = *(const v8bf*)&elo[nloc][kb + 16 + 8 * half];
    v16bf ah, al;
#pragma unroll
    for (int e = 0; e < 8; ++e) {
      ah[e] = a0h[e]; ah[8 + e] = a1h[e];
      al[e] = a0l[e]; al[8 + e] = a1l[e];
    }
    // B fragments: pre-packed, 32 contiguous bytes per lane
    const size_t boff = (size_t)(((kblk * 128 + colb + nloc) * 2 + half)) * 16;
    v16bf bh = *(const v16bf*)(WeHi + boff);
    v16bf bl = *(const v16bf*)(WeLo + boff);

    c = wmma_bf16(ah, bh, c);   // hi*hi
    c = wmma_bf16(al, bh, c);   // lo*hi
    c = wmma_bf16(ah, bl, c);   // hi*lo
  }

  const float aw = Aw[h * 16 + nloc];
  float* lrow = logits + ((size_t)(b * Hh + h) * Nn + i) * Nn + jb;
#pragma unroll
  for (int r = 0; r < 8; ++r) {
    int m = r + 8 * half;
    float v = c[r] + radd[colb + nloc] + p1t[m][colb + nloc];
    v = v > 0.f ? v : 0.01f * v;                 // leaky_relu
    v *= aw;
    v += __shfl_xor(v, 1, 32);                   // reduce 16 mid channels
    v += __shfl_xor(v, 2, 32);
    v += __shfl_xor(v, 4, 32);
    v += __shfl_xor(v, 8, 32);
    if (nloc == 0) lrow[m] = v;                  // Ab cancels in softmax
  }
}

// ---------------------------------------------------------------------------
// Kernel 3: masked softmax over j (row length 512), in place.
// ---------------------------------------------------------------------------
__global__ void softmax_kernel(float* __restrict__ logits,
                               const float* __restrict__ adj) {
  const int i = blockIdx.x, h = blockIdx.y, b = blockIdx.z;
  float* row = logits + ((size_t)(b * Hh + h) * Nn + i) * Nn;
  const float* arow = adj + ((size_t)b * Nn + i) * Nn;
  const int tid = threadIdx.x;
  __shared__ float red[256];

  float x0 = row[tid]       + (arow[tid]       - 1.f) * 1e9f;
  float x1 = row[tid + 256] + (arow[tid + 256] - 1.f) * 1e9f;

  red[tid] = fmaxf(x0, x1);
  __syncthreads();
  for (int s = 128; s > 0; s >>= 1) {
    if (tid < s) red[tid] = fmaxf(red[tid], red[tid + s]);
    __syncthreads();
  }
  float mx = red[0];
  __syncthreads();

  float e0 = __expf(x0 - mx), e1 = __expf(x1 - mx);
  red[tid] = e0 + e1;
  __syncthreads();
  for (int s = 128; s > 0; s >>= 1) {
    if (tid < s) red[tid] += red[tid + s];
    __syncthreads();
  }
  float inv = 1.f / red[0];
  row[tid]       = e0 * inv;
  row[tid + 256] = e1 * inv;
}

// ---------------------------------------------------------------------------
// Kernel 4: out = relu( coefs @ values + skip )  (fp32 WMMA, small)
// ---------------------------------------------------------------------------
__global__ void out_kernel(const float* __restrict__ coefs,
                           const float* __restrict__ V,
                           const float* __restrict__ S,
                           float* __restrict__ out) {
  const int it = blockIdx.x, b = blockIdx.y;
  const int ib = it * 16;
  const int tid = threadIdx.x;
  const int h = tid >> 5, lane = tid & 31;
  const int nloc = lane & 15, half = lane >> 4;
  const int colb = h * 16;

  const float* crow = coefs + (size_t)(b * Hh + h) * Nn * Nn;
  v8f c = {};
  for (int k = 0; k < Nn; k += 4) {
    int ka = k + half * 2;
    v2f a, bf;
    a.x  = crow[(size_t)(ib + nloc) * Nn + ka];
    a.y  = crow[(size_t)(ib + nloc) * Nn + ka + 1];
    bf.x = V[((size_t)b * Nn + ka) * 128 + colb + nloc];
    bf.y = V[((size_t)b * Nn + ka + 1) * 128 + colb + nloc];
    c = wmma_f32(a, bf, c);
  }
#pragma unroll
  for (int r = 0; r < 8; ++r) {
    int irow = ib + r + 8 * half;
    float v = c[r] + S[((size_t)b * Nn + irow) * 128 + colb + nloc];
    out[((size_t)b * Nn + irow) * 128 + colb + nloc] = fmaxf(v, 0.f);
  }
}

// ---------------------------------------------------------------------------
extern "C" void kernel_launch(void* const* d_in, const int* in_sizes, int n_in,
                              void* d_out, int out_size, void* d_ws, size_t ws_size,
                              hipStream_t stream) {
  const float* node   = (const float*)d_in[0];
  const float* edge   = (const float*)d_in[1];
  const float* gfts   = (const float*)d_in[2];
  const float* adj    = (const float*)d_in[3];
  const float* hidden = (const float*)d_in[4];
  const float* Wm = (const float*)d_in[5];  const float* bm = (const float*)d_in[6];
  const float* Ws = (const float*)d_in[7];  const float* bs = (const float*)d_in[8];
  const float* W1 = (const float*)d_in[9];  const float* b1 = (const float*)d_in[10];
  const float* W2 = (const float*)d_in[11]; const float* b2 = (const float*)d_in[12];
  const float* We = (const float*)d_in[13]; const float* be = (const float*)d_in[14];
  const float* Wg = (const float*)d_in[15]; const float* bg = (const float*)d_in[16];
  const float* Aw = (const float*)d_in[17];
  // Ab (d_in[18]) cancels in the softmax -> unused.

  float* ws = (float*)d_ws;
  const size_t PROJ = (size_t)Bb * Nn * 128;     // 262144
  float* P1 = ws;
  float* P2 = P1 + PROJ;
  float* Vv = P2 + PROJ;
  float* Ss = Vv + PROJ;
  float* G  = Ss + PROJ;                         // 512
  float* LG = G + Bb * 128;                      // B*H*N*N = 8.4M floats
  __bf16* WeHi = (__bf16*)(LG + (size_t)Bb * Hh * Nn * Nn);  // 16384 bf16
  __bf16* WeLo = WeHi + 16384;

  graph_proj_kernel<<<2, 256, 0, stream>>>(gfts, Wg, bg, G);

  we_pack_kernel<<<64, 256, 0, stream>>>(We, WeHi, WeLo);

  proj_z_kernel<<<(Bb * Nn) / 16, 256, 0, stream>>>(
      node, hidden, Wm, bm, Ws, bs, W1, b1, W2, b2, Vv, Ss, P1, P2);

  edge_logits_kernel<<<dim3(Nn / 16, Nn, Bb), 256, 0, stream>>>(
      edge, WeHi, WeLo, be, P1, P2, G, Aw, LG);

  softmax_kernel<<<dim3(Nn, Hh, Bb), 256, 0, stream>>>(LG, adj);

  out_kernel<<<dim3(Nn / 16, Bb), 256, 0, stream>>>(LG, Vv, Ss, (float*)d_out);
}
